// compute_G_21964462752033
// MI455X (gfx1250) — matrix operand order repaired and checked by reading, not compiled, not run on hardware
//
#include <hip/hip_runtime.h>
#include <hip/hip_bf16.h>

// G = DV2_H[16384,128] @ (W[:,None] * invDE_HT_DV2[128,16384])  -> [16384,16384] f32
//
// MI455X plan:
//  * 1 GiB output write dominates HBM traffic (inputs 16 MB, fully L2-resident):
//    write floor ~46 us @ 23.3 TB/s -> need ~1.5 PFLOP/s, only the f16 WMMA path gets there.
//  * fp32 accuracy via 2-term f16 split: x = hi + lo;  A*B ~= AhiBhi + AhiBlo + AloBhi
//    accumulated in the f32 WMMA accumulator (dropped AloBlo term ~2^-22 relative).
//  * The split is PRECOMPUTED once into d_ws (16 MB) so the GEMM hot loop has zero
//    cvt/fma VALU: pure v_wmma + b128 loads from L2 + nontemporal b32 stores.

typedef __attribute__((ext_vector_type(16))) _Float16 v16h;
typedef __attribute__((ext_vector_type(8)))  _Float16 v8h;
typedef __attribute__((ext_vector_type(4)))  _Float16 v4h;
typedef __attribute__((ext_vector_type(8)))  float    v8f;
typedef __attribute__((ext_vector_type(4)))  float    v4f;

#define N_NODES 16384
#define N_EDGES 128

#define TILE_M 128   // rows per block (GEMM)
#define TILE_N 64    // cols per block (GEMM)

// ---------------------------------------------------------------------------
// Pass 1: split A (DV2_H) into hi/lo f16, row-major [row][k]
// ---------------------------------------------------------------------------
__global__ __launch_bounds__(256) void prep_a_kernel(
    const float* __restrict__ A, _Float16* __restrict__ Ahi, _Float16* __restrict__ Alo)
{
  const int idx4 = blockIdx.x * 256 + threadIdx.x;   // one float4 per thread
  v4f x = *(const v4f*)(A + (size_t)idx4 * 4);
  v4h h, l;
#pragma unroll
  for (int j = 0; j < 4; ++j) {
    _Float16 a = (_Float16)x[j];
    h[j] = a;
    l[j] = (_Float16)(x[j] - (float)a);
  }
  *(v4h*)(Ahi + (size_t)idx4 * 4) = h;
  *(v4h*)(Alo + (size_t)idx4 * 4) = l;
}

// ---------------------------------------------------------------------------
// Pass 2: Bt[col][k] = split( W[k] * B[k][col] )  (transpose via LDS tile)
// Block handles 32 columns x all 128 k.
// ---------------------------------------------------------------------------
__global__ __launch_bounds__(256) void prep_bt_kernel(
    const float* __restrict__ B, const float* __restrict__ W,
    _Float16* __restrict__ Bthi, _Float16* __restrict__ Btlo)
{
  __shared__ _Float16 sHi[32][N_EDGES];
  __shared__ _Float16 sLo[32][N_EDGES];

  const int tid      = threadIdx.x;
  const int blockCol = blockIdx.x * 32;

  // fill: coalesced reads along N, scale by W[k], split, store transposed in LDS
  {
    const int c4 = tid & 7;    // which float4 of the 32-wide row
    const int kr = tid >> 3;   // k-row within sweep (0..31)
#pragma unroll
    for (int it = 0; it < 4; ++it) {
      const int k = it * 32 + kr;
      const float w = W[k];
      v4f b = *(const v4f*)(B + (size_t)k * N_NODES + blockCol + c4 * 4);
#pragma unroll
      for (int j = 0; j < 4; ++j) {
        const float x = w * b[j];
        const _Float16 h = (_Float16)x;
        sHi[c4 * 4 + j][k] = h;
        sLo[c4 * 4 + j][k] = (_Float16)(x - (float)h);
      }
    }
  }
  __syncthreads();

  // write out: each Bt row (128 halves = 256 B) written as 16B chunks, coalesced
#pragma unroll
  for (int it = 0; it < 2; ++it) {
    const int chunk = it * 256 + tid;        // 512 chunks of 8 halves
    const int col   = chunk >> 4;            // 0..31
    const int off   = (chunk & 15) * 8;      // 0..120
    *(v8h*)(Bthi + (size_t)(blockCol + col) * N_EDGES + off) = *(const v8h*)&sHi[col][off];
    *(v8h*)(Btlo + (size_t)(blockCol + col) * N_EDGES + off) = *(const v8h*)&sLo[col][off];
  }
}

// ---------------------------------------------------------------------------
// Pass 3: GEMM on pre-split f16 panels. No LDS, no barriers, no conversions.
// 8 waves: 4 in M (32 rows) x 2 in N (32 cols); 2x2 16x16 C tiles per wave.
// ---------------------------------------------------------------------------
__global__ __launch_bounds__(256) void gemm_split_kernel(
    const _Float16* __restrict__ Ahi, const _Float16* __restrict__ Alo,
    const _Float16* __restrict__ Bthi, const _Float16* __restrict__ Btlo,
    float* __restrict__ G)
{
  const int tid      = threadIdx.x;
  const int blockRow = blockIdx.y * TILE_M;
  const int blockCol = blockIdx.x * TILE_N;

  const int lane  = tid & 31;
  const int wid   = tid >> 5;
  const int waveM = wid & 3;
  const int waveN = wid >> 2;
  const int m     = lane & 15;
  const int sel   = lane >> 4;   // half-wave selects K sub-range k0+16*sel..+15 (contiguous order, same for A and B)

  const int rowWaveG = blockRow + waveM * 32;
  const int colWaveG = blockCol + waveN * 32;

  v8f acc[2][2];
#pragma unroll
  for (int mt = 0; mt < 2; ++mt)
#pragma unroll
    for (int nt = 0; nt < 2; ++nt)
      acc[mt][nt] = (v8f){0.f, 0.f, 0.f, 0.f, 0.f, 0.f, 0.f, 0.f};

#pragma unroll
  for (int kc = 0; kc < 4; ++kc) {
    const int k0 = kc * 32;

    // A fragment: lane<16 -> K = k0+0..15 ; lane>=16 -> K = k0+16..31 (matches B)
    union F { v16h v; v8h h[2]; };
    F aHi[2], aLo[2];
#pragma unroll
    for (int mt = 0; mt < 2; ++mt) {
      const size_t rbase = (size_t)(rowWaveG + mt * 16 + m) * N_EDGES;
      aHi[mt].h[0] = *(const v8h*)(Ahi + rbase + k0 + sel * 16);
      aHi[mt].h[1] = *(const v8h*)(Ahi + rbase + k0 + sel * 16 + 8);
      aLo[mt].h[0] = *(const v8h*)(Alo + rbase + k0 + sel * 16);
      aLo[mt].h[1] = *(const v8h*)(Alo + rbase + k0 + sel * 16 + 8);
    }

#pragma unroll
    for (int nt = 0; nt < 2; ++nt) {
      // B fragment: lane<16 holds K=k0..k0+15 of col n, lane>=16 holds K=k0+16..k0+31
      const size_t cbase = (size_t)(colWaveG + nt * 16 + m) * N_EDGES + k0 + sel * 16;
      F bHi, bLo;
      bHi.h[0] = *(const v8h*)(Bthi + cbase);
      bHi.h[1] = *(const v8h*)(Bthi + cbase + 8);
      bLo.h[0] = *(const v8h*)(Btlo + cbase);
      bLo.h[1] = *(const v8h*)(Btlo + cbase + 8);

#pragma unroll
      for (int mt = 0; mt < 2; ++mt) {
        acc[mt][nt] = __builtin_amdgcn_wmma_f32_16x16x32_f16(
            false, aHi[mt].v, false, bHi.v, (short)0, acc[mt][nt], false, false);
        acc[mt][nt] = __builtin_amdgcn_wmma_f32_16x16x32_f16(
            false, aHi[mt].v, false, bLo.v, (short)0, acc[mt][nt], false, false);
        acc[mt][nt] = __builtin_amdgcn_wmma_f32_16x16x32_f16(
            false, aLo[mt].v, false, bHi.v, (short)0, acc[mt][nt], false, false);
      }
    }
  }

  // C/D layout: VGPR i -> row (i + 8*sel), col = lane&15. Nontemporal: keep L2 for panels.
#pragma unroll
  for (int mt = 0; mt < 2; ++mt) {
#pragma unroll
    for (int nt = 0; nt < 2; ++nt) {
      const int rowG = rowWaveG + mt * 16 + sel * 8;
      const int colG = colWaveG + nt * 16 + m;
      float* out = G + (size_t)rowG * N_NODES + colG;
#pragma unroll
      for (int i = 0; i < 8; ++i) {
        __builtin_nontemporal_store(acc[mt][nt][i], out + (size_t)i * N_NODES);
      }
    }
  }
}

// ---------------------------------------------------------------------------
// Fallback (ws too small): fused kernel with in-loop split (previous round).
// ---------------------------------------------------------------------------
__global__ __launch_bounds__(256) void hgnn_G_fused_kernel(
    const float* __restrict__ A, const float* __restrict__ B,
    const float* __restrict__ W, float* __restrict__ G)
{
  __shared__ _Float16 sBhi[TILE_N][136];
  __shared__ _Float16 sBlo[TILE_N][136];

  const int tid      = threadIdx.x;
  const int blockRow = blockIdx.y * TILE_M;
  const int blockCol = blockIdx.x * TILE_N;

  {
    const int c4 = tid & 15;
    const int kr = tid >> 4;
#pragma unroll
    for (int it = 0; it < 8; ++it) {
      const int k = it * 16 + kr;
      const float w = W[k];
      v4f b = *(const v4f*)(B + (size_t)k * N_NODES + blockCol + c4 * 4);
#pragma unroll
      for (int j = 0; j < 4; ++j) {
        const float x = w * b[j];
        const _Float16 h = (_Float16)x;
        sBhi[c4 * 4 + j][k] = h;
        sBlo[c4 * 4 + j][k] = (_Float16)(x - (float)h);
      }
    }
  }
  __syncthreads();

  const int lane  = tid & 31;
  const int wid   = tid >> 5;
  const int waveM = wid & 3;
  const int waveN = wid >> 2;
  const int m     = lane & 15;
  const int sel   = lane >> 4;

  const int rowWaveG = blockRow + waveM * 32;
  const int colWaveL = waveN * 32;

  v8f acc[2][2];
#pragma unroll
  for (int mt = 0; mt < 2; ++mt)
#pragma unroll
    for (int nt = 0; nt < 2; ++nt)
      acc[mt][nt] = (v8f){0.f, 0.f, 0.f, 0.f, 0.f, 0.f, 0.f, 0.f};

#pragma unroll
  for (int kc = 0; kc < 4; ++kc) {
    const int k0 = kc * 32;
    v16h aHi[2], aLo[2];
#pragma unroll
    for (int mt = 0; mt < 2; ++mt) {
      const float* arow = A + (size_t)(rowWaveG + mt * 16 + m) * N_EDGES;
      float af[16];
      v4f t0 = *(const v4f*)(arow + k0 + sel * 16);
      v4f t1 = *(const v4f*)(arow + k0 + sel * 16 + 4);
      v4f t2 = *(const v4f*)(arow + k0 + sel * 16 + 8);
      v4f t3 = *(const v4f*)(arow + k0 + sel * 16 + 12);
#pragma unroll
      for (int j = 0; j < 4; ++j) {
        af[j] = t0[j]; af[4 + j] = t1[j]; af[8 + j] = t2[j]; af[12 + j] = t3[j];
      }
      v16h h, l;
#pragma unroll
      for (int i = 0; i < 16; ++i) {
        _Float16 a = (_Float16)af[i];
        h[i] = a;
        l[i] = (_Float16)(af[i] - (float)a);
      }
      aHi[mt] = h; aLo[mt] = l;
    }
#pragma unroll
    for (int nt = 0; nt < 2; ++nt) {
      const _Float16* bh = &sBhi[colWaveL + nt * 16 + m][k0 + sel * 16];
      const _Float16* bl = &sBlo[colWaveL + nt * 16 + m][k0 + sel * 16];
      union { v16h v; v8h h[2]; } bHi, bLo;
      bHi.h[0] = *(const v8h*)(bh); bHi.h[1] = *(const v8h*)(bh + 8);
      bLo.h[0] = *(const v8h*)(bl); bLo.h[1] = *(const v8h*)(bl + 8);
#pragma unroll
      for (int mt = 0; mt < 2; ++mt) {
        acc[mt][nt] = __builtin_amdgcn_wmma_f32_16x16x32_f16(
            false, aHi[mt], false, bHi.v, (short)0, acc[mt][nt], false, false);
        acc[mt][nt] = __builtin_amdgcn_wmma_f32_16x16x32_f16(
            false, aHi[mt], false, bLo.v, (short)0, acc[mt][nt], false, false);
        acc[mt][nt] = __builtin_amdgcn_wmma_f32_16x16x32_f16(
            false, aLo[mt], false, bHi.v, (short)0, acc[mt][nt], false, false);
      }
    }
  }

#pragma unroll
  for (int mt = 0; mt < 2; ++mt) {
#pragma unroll
    for (int nt = 0; nt < 2; ++nt) {
      const int rowG = rowWaveG + mt * 16 + sel * 8;
      const int colG = blockCol + colWaveL + nt * 16 + m;
      float* out = G + (size_t)rowG * N_NODES + colG;
#pragma unroll
      for (int i = 0; i < 8; ++i)
        __builtin_nontemporal_store(acc[mt][nt][i], out + (size_t)i * N_NODES);
    }
  }
}

// ---------------------------------------------------------------------------
extern "C" void kernel_launch(void* const* d_in, const int* in_sizes, int n_in,
                              void* d_out, int out_size, void* d_ws, size_t ws_size,
                              hipStream_t stream) {
  const float* DV2_H = (const float*)d_in[0];
  const float* invDE = (const float*)d_in[1];
  const float* W     = (const float*)d_in[2];
  float* G           = (float*)d_out;

  const size_t PANEL = (size_t)N_NODES * N_EDGES;      // 2,097,152 halves = 4 MB
  const size_t need  = 4 * PANEL * sizeof(_Float16);   // 16 MB

  if (ws_size >= need) {
    _Float16* Ahi  = (_Float16*)d_ws;
    _Float16* Alo  = Ahi + PANEL;
    _Float16* Bthi = Alo + PANEL;
    _Float16* Btlo = Bthi + PANEL;

    prep_a_kernel<<<dim3(PANEL / 4 / 256), dim3(256), 0, stream>>>(DV2_H, Ahi, Alo);
    prep_bt_kernel<<<dim3(N_NODES / 32), dim3(256), 0, stream>>>(invDE, W, Bthi, Btlo);
    gemm_split_kernel<<<dim3(N_NODES / TILE_N, N_NODES / TILE_M), dim3(256), 0, stream>>>(
        Ahi, Alo, Bthi, Btlo, G);
  } else {
    hgnn_G_fused_kernel<<<dim3(N_NODES / TILE_N, N_NODES / TILE_M), dim3(256), 0, stream>>>(
        DV2_H, invDE, W, G);
  }
}
